// PatchCore_9457517985864
// MI455X (gfx1250) — matrix operand order repaired and checked by reading, not compiled, not run on hardware
//
#include <hip/hip_runtime.h>

typedef __attribute__((ext_vector_type(8)))  float   v8f;
typedef __attribute__((ext_vector_type(16))) __bf16  v16bf;

#define GN 8192      // feature rows (B*H*W)
#define GM 16384     // memory bank rows
#define GC 1536      // channels
#define GB 32        // batch
#define IM 256       // image size
#define KS 25        // gaussian taps
#define KHALF 12

#define TROWS 128
#define TCOLS 128
#define TK 64
#define NKCH (GC / TK)                 // 24 K-chunks
#define SPLIT 8
#define COLS_PER_WG (GM / SPLIT)       // 2048
#define NCHUNK (COLS_PER_WG / TCOLS)   // 16

// ---------- helpers ----------
__device__ inline unsigned short f2bf_rne(float x) {
  unsigned u = __float_as_uint(x);
  u += 0x7FFFu + ((u >> 16) & 1u);
  return (unsigned short)(u >> 16);
}
__device__ inline float bf2f(unsigned short h) {
  return __uint_as_float(((unsigned)h) << 16);
}
__device__ inline unsigned long long shfl_xor_u64(unsigned long long v, int m) {
  unsigned lo = (unsigned)v, hi = (unsigned)(v >> 32);
  lo = __shfl_xor(lo, m, 32);
  hi = __shfl_xor(hi, m, 32);
  return ((unsigned long long)hi << 32) | lo;
}

// bank-conflict-avoiding swizzles (keep 8/16-element fragment chunks contiguous)
__device__ inline int AIDX(int r, int k) { return r * TK    + (k ^ ((r & 7) << 3)); }
__device__ inline int BIDX(int k, int n) { return k * TCOLS + (n ^ ((k & 7) << 4)); }

union FragBF { uint4 q[2]; v16bf v; };

// ---------- fused distance GEMM + row min/argmin (software-pipelined) ----------
__global__ __launch_bounds__(256) void knn_min_gemm(
    const float* __restrict__ F, const float* __restrict__ Mb,
    const float* __restrict__ fnorm, const float* __restrict__ mnorm,
    unsigned long long* __restrict__ keys)
{
  __shared__ __align__(16) unsigned short Ahi[TROWS * TK];
  __shared__ __align__(16) unsigned short Alo[TROWS * TK];
  __shared__ __align__(16) unsigned short Bhi[TK * TCOLS];
  __shared__ __align__(16) unsigned short Blo[TK * TCOLS];

  const int tid  = threadIdx.x;
  const int lane = tid & 31;
  const int wave = tid >> 5;
  const int wrow = wave >> 1;   // 0..3 -> 32-row band
  const int wcol = wave & 1;    // 0..1 -> 64-col band
  const int l15  = lane & 15;
  const int lhv  = lane >> 4;   // 0/1

  const int rowBlk   = blockIdx.x >> 3;
  const int split    = blockIdx.x & 7;
  const int rowBase  = rowBlk * TROWS;
  const int colBase0 = split * COLS_PER_WG;

  // cooperative-load coordinates for this thread (16 threads share a row)
  const int ldR  = tid >> 4;        // 0..15 base row; +16 per step of i
  const int ldKQ = tid & 15;        // float4 slot along K

  const v8f vz = {0.f,0.f,0.f,0.f,0.f,0.f,0.f,0.f};

  float fn[2][8];
  #pragma unroll
  for (int tr = 0; tr < 2; ++tr)
    #pragma unroll
    for (int v = 0; v < 8; ++v)
      fn[tr][v] = fnorm[rowBase + wrow*32 + tr*16 + v + 8*lhv];

  unsigned long long runmin[2][8];
  #pragma unroll
  for (int tr = 0; tr < 2; ++tr)
    #pragma unroll
    for (int v = 0; v < 8; ++v) runmin[tr][v] = ~0ull;

  for (int cc = 0; cc < NCHUNK; ++cc) {
    const int col0 = colBase0 + cc * TCOLS;
    v8f acc[2][4];
    #pragma unroll
    for (int i = 0; i < 2; ++i)
      #pragma unroll
      for (int j = 0; j < 4; ++j) acc[i][j] = vz;

    // ---- prologue: stage first K-chunk in registers ----
    float4 ra[8], rb[8];
    #pragma unroll
    for (int i = 0; i < 8; ++i) {
      int r = ldR + i * 16;
      ra[i] = *(const float4*)&F [(size_t)(rowBase + r) * GC + ldKQ * 4];
      rb[i] = *(const float4*)&Mb[(size_t)(col0    + r) * GC + ldKQ * 4];
    }

    for (int kbi = 0; kbi < NKCH; ++kbi) {
      __syncthreads();   // LDS free from previous compute

      // ---- drain staged registers: f32 -> (bf16 hi, lo) split into LDS ----
      #pragma unroll
      for (int i = 0; i < 8; ++i) {
        int r = ldR + i * 16;
        float av[4] = {ra[i].x, ra[i].y, ra[i].z, ra[i].w};
        float bv[4] = {rb[i].x, rb[i].y, rb[i].z, rb[i].w};
        ushort4 ah4, al4;
        unsigned short hs[4], ls[4];
        #pragma unroll
        for (int j = 0; j < 4; ++j) {
          hs[j] = f2bf_rne(av[j]);
          ls[j] = f2bf_rne(av[j] - bf2f(hs[j]));
        }
        ah4.x = hs[0]; ah4.y = hs[1]; ah4.z = hs[2]; ah4.w = hs[3];
        al4.x = ls[0]; al4.y = ls[1]; al4.z = ls[2]; al4.w = ls[3];
        // packed 64-bit stores; (kq*4)^swizzle stays 4-aligned & contiguous
        *(ushort4*)&Ahi[AIDX(r, ldKQ * 4)] = ah4;
        *(ushort4*)&Alo[AIDX(r, ldKQ * 4)] = al4;
        #pragma unroll
        for (int j = 0; j < 4; ++j) {
          unsigned short g = f2bf_rne(bv[j]);
          Bhi[AIDX(r, ldKQ*4 + j)] = g;
          Blo[AIDX(r, ldKQ*4 + j)] = f2bf_rne(bv[j] - bf2f(g));
        }
      }
      __syncthreads();

      // ---- issue next chunk's global loads BEFORE compute (latency hiding) ----
      if (kbi + 1 < NKCH) {
        int kb2 = (kbi + 1) * TK;
        #pragma unroll
        for (int i = 0; i < 8; ++i) {
          int r = ldR + i * 16;
          ra[i] = *(const float4*)&F [(size_t)(rowBase + r) * GC + kb2 + ldKQ * 4];
          rb[i] = *(const float4*)&Mb[(size_t)(col0    + r) * GC + kb2 + ldKQ * 4];
        }
      }

      // ---- WMMA compute on current chunk ----
      #pragma unroll
      for (int ks = 0; ks < 2; ++ks) {
        FragBF ah[2], al[2];
        #pragma unroll
        for (int tr = 0; tr < 2; ++tr) {
          int r  = wrow*32 + tr*16 + l15;
          int k0 = ks*32 + 8*lhv;
          ah[tr].q[0] = *(const uint4*)&Ahi[AIDX(r, k0)];
          ah[tr].q[1] = *(const uint4*)&Ahi[AIDX(r, k0 + 16)];
          al[tr].q[0] = *(const uint4*)&Alo[AIDX(r, k0)];
          al[tr].q[1] = *(const uint4*)&Alo[AIDX(r, k0 + 16)];
        }
        #pragma unroll
        for (int tc = 0; tc < 4; ++tc) {
          int k  = ks*32 + 8*lhv;
          int n0 = wcol*64 + tc*16 + l15;
          FragBF bh, bl;
          int bbase = AIDX(n0, k);
          bh.q[0] = *(const uint4*)&Bhi[bbase];
          bh.q[1] = *(const uint4*)&Bhi[AIDX(n0, k + 16)];
          bl.q[0] = *(const uint4*)&Blo[bbase];
          bl.q[1] = *(const uint4*)&Blo[AIDX(n0, k + 16)];
          #pragma unroll
          for (int tr = 0; tr < 2; ++tr) {
            v8f c = acc[tr][tc];
            // bf16 split compensation: hi*hi + hi*lo + lo*hi ~= fp32 dot
            c = __builtin_amdgcn_wmma_f32_16x16x32_bf16(false, ah[tr].v, false, bh.v, (short)0, c, false, false);
            c = __builtin_amdgcn_wmma_f32_16x16x32_bf16(false, ah[tr].v, false, bl.v, (short)0, c, false, false);
            c = __builtin_amdgcn_wmma_f32_16x16x32_bf16(false, al[tr].v, false, bh.v, (short)0, c, false, false);
            acc[tr][tc] = c;
          }
        }
      }
    }

    // fold this 128-col chunk into running per-row minima
    #pragma unroll
    for (int tc = 0; tc < 4; ++tc) {
      int col = col0 + wcol*64 + tc*16 + l15;
      float mn = mnorm[col];
      #pragma unroll
      for (int tr = 0; tr < 2; ++tr) {
        #pragma unroll
        for (int v = 0; v < 8; ++v) {
          float d2 = fn[tr][v] + mn - 2.0f * acc[tr][tc][v];
          d2 = fmaxf(d2, 0.0f);
          unsigned long long key =
              ((unsigned long long)__float_as_uint(d2) << 32) | (unsigned)col;
          runmin[tr][v] = runmin[tr][v] < key ? runmin[tr][v] : key;
        }
      }
    }
  }

  // reduce across the 16 lanes that hold different columns of each row
  #pragma unroll
  for (int tr = 0; tr < 2; ++tr) {
    #pragma unroll
    for (int v = 0; v < 8; ++v) {
      unsigned long long k = runmin[tr][v];
      #pragma unroll
      for (int off = 1; off < 16; off <<= 1) {
        unsigned long long o = shfl_xor_u64(k, off);
        k = k < o ? k : o;
      }
      if (l15 == 0) {
        int row = rowBase + wrow*32 + tr*16 + v + 8*lhv;
        atomicMin(&keys[row], k);
      }
    }
  }
}

// ---------- small kernels ----------
__global__ void init_keys(unsigned long long* keys) {
  int i = blockIdx.x*256 + threadIdx.x;
  if (i < GN) keys[i] = ~0ull;
}

__global__ void row_norms(const float* __restrict__ X, float* __restrict__ out) {
  __shared__ float red[256];
  int row = blockIdx.x;
  const float* p = X + (size_t)row * GC;
  float s = 0.f;
  for (int k = threadIdx.x; k < GC; k += 256) { float x = p[k]; s += x*x; }
  red[threadIdx.x] = s; __syncthreads();
  for (int w = 128; w > 0; w >>= 1) {
    if (threadIdx.x < w) red[threadIdx.x] += red[threadIdx.x + w];
    __syncthreads();
  }
  if (threadIdx.x == 0) out[row] = red[0];
}

__global__ void finalize_patch(const unsigned long long* __restrict__ keys,
                               float* __restrict__ pscore, float* __restrict__ sstar,
                               int* __restrict__ midx) {
  __shared__ unsigned long long red[256];
  int b = blockIdx.x, t = threadIdx.x;
  unsigned long long k = keys[b*256 + t];
  float dist = sqrtf(__uint_as_float((unsigned)(k >> 32)));
  pscore[b*256 + t] = dist;
  // max with first-index tie-break: pack (255 - t) low
  unsigned long long mk =
      ((unsigned long long)__float_as_uint(dist) << 32) | (unsigned)(255 - t);
  red[t] = mk; __syncthreads();
  for (int w = 128; w > 0; w >>= 1) {
    if (t < w) red[t] = red[t] > red[t + w] ? red[t] : red[t + w];
    __syncthreads();
  }
  if (t == 0) {
    unsigned long long mx = red[0];
    int tstar = 255 - (int)(mx & 0xFFFFFFFFu);
    sstar[b] = __uint_as_float((unsigned)(mx >> 32));
    midx[b]  = (int)(keys[b*256 + tstar] & 0xFFFFFFFFu);
  }
}

__global__ void mstar_dists(const float* __restrict__ Mb, const float* __restrict__ mnorm,
                            const int* __restrict__ midx, float* __restrict__ md) {
  int gid = blockIdx.x*256 + threadIdx.x;
  int b = gid >> 14;
  int j = gid & (GM - 1);
  if (b >= GB) return;
  int ms = midx[b];
  const float4* a = (const float4*)(Mb + (size_t)ms * GC);
  const float4* c = (const float4*)(Mb + (size_t)j  * GC);
  float dot = 0.f;
  for (int k = 0; k < GC/4; ++k) {
    float4 x = a[k], y = c[k];
    dot += x.x*y.x + x.y*y.y + x.z*y.z + x.w*y.w;
  }
  float d2 = mnorm[ms] + mnorm[j] - 2.f * dot;
  md[(size_t)b * GM + j] = sqrtf(fmaxf(d2, 0.f));
}

__global__ void topk_score(const float* __restrict__ md, const float* __restrict__ sstar,
                           float* __restrict__ out) {
  __shared__ float cand[256 * 10];
  int b = blockIdx.x, t = threadIdx.x;
  float loc[10];
  #pragma unroll
  for (int i = 0; i < 10; ++i) loc[i] = 3.0e38f;
  for (int j = t; j < GM; j += 256) {
    float d = md[(size_t)b * GM + j];
    if (d < loc[9]) {
      int p = 9;
      while (p > 0 && loc[p-1] > d) { loc[p] = loc[p-1]; --p; }
      loc[p] = d;
    }
  }
  #pragma unroll
  for (int i = 0; i < 10; ++i) cand[t*10 + i] = loc[i];
  __syncthreads();
  if (t == 0) {
    float best[10];
    for (int s = 0; s < 10; ++s) {
      float mv = 3.0e38f; int mi = 0;
      for (int i = 0; i < 2560; ++i) if (cand[i] < mv) { mv = cand[i]; mi = i; }
      best[s] = mv; cand[mi] = 3.0e38f;
    }
    float sum = 0.f;
    for (int i = 1; i < 10; ++i) sum += expf(best[i]);
    float w = 1.0f - expf(best[1]) / sum;   // drop self-distance best[0]
    out[b] = w * sstar[b];
  }
}

__global__ void upsample_bilinear(const float* __restrict__ ps, float* __restrict__ up) {
  int gid = blockIdx.x*256 + threadIdx.x;   // GB*IM*IM
  int b = gid >> 16;
  int y = (gid >> 8) & 255;
  int x = gid & 255;
  float fy = (y + 0.5f) * (1.0f/16.0f) - 0.5f;
  float fx = (x + 0.5f) * (1.0f/16.0f) - 0.5f;
  float y0f = floorf(fy), x0f = floorf(fx);
  int y0 = (int)y0f, x0 = (int)x0f;
  float wy = fy - y0f, wx = fx - x0f;
  int y0c = min(max(y0, 0), 15), y1c = min(max(y0 + 1, 0), 15);
  int x0c = min(max(x0, 0), 15), x1c = min(max(x0 + 1, 0), 15);
  const float* p = ps + b * 256;
  float v00 = p[y0c*16 + x0c], v01 = p[y0c*16 + x1c];
  float v10 = p[y1c*16 + x0c], v11 = p[y1c*16 + x1c];
  up[gid] = (1.f-wy)*((1.f-wx)*v00 + wx*v01) + wy*((1.f-wx)*v10 + wx*v11);
}

__device__ inline void gauss_weights(float* g) {
  float s = 0.f;
  #pragma unroll
  for (int i = 0; i < KS; ++i) {
    float x = (float)(i - KHALF) * 0.25f;     // /sigma, sigma=4
    g[i] = expf(-0.5f * x * x);
    s += g[i];
  }
  float inv = 1.0f / s;
  #pragma unroll
  for (int i = 0; i < KS; ++i) g[i] *= inv;
}

__global__ void blur_h(const float* __restrict__ in, float* __restrict__ out) {
  float g[KS]; gauss_weights(g);
  int gid = blockIdx.x*256 + threadIdx.x;
  int x = gid & 255;
  int rowBase = gid - x;
  float acc = 0.f;
  #pragma unroll
  for (int i = 0; i < KS; ++i) {
    int xx = x + i - KHALF;
    if (xx >= 0 && xx < 256) acc += g[i] * in[rowBase + xx];
  }
  out[gid] = acc;
}

__global__ void blur_v(const float* __restrict__ in, float* __restrict__ out) {
  float g[KS]; gauss_weights(g);
  int gid = blockIdx.x*256 + threadIdx.x;
  int b = gid >> 16;
  int y = (gid >> 8) & 255;
  int x = gid & 255;
  float acc = 0.f;
  #pragma unroll
  for (int i = 0; i < KS; ++i) {
    int yy = y + i - KHALF;
    if (yy >= 0 && yy < 256) acc += g[i] * in[(b << 16) + yy*256 + x];
  }
  out[gid] = acc;
}

// ---------- launch ----------
extern "C" void kernel_launch(void* const* d_in, const int* in_sizes, int n_in,
                              void* d_out, int out_size, void* d_ws, size_t ws_size,
                              hipStream_t stream) {
  const float* F  = (const float*)d_in[0];
  const float* Mb = (const float*)d_in[1];
  float* out = (float*)d_out;   // [0..32): scores, [32..): amap (32x256x256)

  char* ws = (char*)d_ws;
  unsigned long long* keys = (unsigned long long*)(ws + 0x000000); // 64 KB
  float* fnorm  = (float*)(ws + 0x010000);                         // 32 KB
  float* mnorm  = (float*)(ws + 0x018000);                         // 64 KB
  float* pscore = (float*)(ws + 0x028000);                         // 32 KB
  float* sstar  = (float*)(ws + 0x030000);
  int*   midx   = (int*)  (ws + 0x030100);
  float* md     = (float*)(ws + 0x031000);                         // 2 MB
  float* up     = (float*)(ws + 0x240000);                         // 8 MB
  float* tmp    = (float*)(ws + 0xA40000);                         // 8 MB

  init_keys<<<GN/256, 256, 0, stream>>>(keys);
  row_norms<<<GN, 256, 0, stream>>>(F, fnorm);
  row_norms<<<GM, 256, 0, stream>>>(Mb, mnorm);
  knn_min_gemm<<<(GN/TROWS)*SPLIT, 256, 0, stream>>>(F, Mb, fnorm, mnorm, keys);
  finalize_patch<<<GB, 256, 0, stream>>>(keys, pscore, sstar, midx);
  mstar_dists<<<(GB*GM)/256, 256, 0, stream>>>(Mb, mnorm, midx, md);
  topk_score<<<GB, 256, 0, stream>>>(md, sstar, out);
  upsample_bilinear<<<(GB*IM*IM)/256, 256, 0, stream>>>(pscore, up);
  blur_h<<<(GB*IM*IM)/256, 256, 0, stream>>>(up, tmp);
  blur_v<<<(GB*IM*IM)/256, 256, 0, stream>>>(tmp, out + GB);
}
